// CorrelationLayer_4114578669816
// MI455X (gfx1250) — matrix operand order repaired and checked by reading, not compiled, not run on hardware
//
#include <hip/hip_runtime.h>

typedef __bf16 bf16_t;
typedef __attribute__((ext_vector_type(16))) __bf16 v16bf;
typedef __attribute__((ext_vector_type(8)))  __bf16 v8bf;
typedef __attribute__((ext_vector_type(8)))  float  v8f;

#define C_DIM 128
#define H_DIM 384
#define W_DIM 512
#define MAXD  3
#define BH    2                 // output h-rows per block
#define KP    136               // padded channel stride in LDS (bank spread)
#define YROWS (BH + 2*MAXD)     // 8 y rows staged
#define YCOLS (16 + 2*MAXD)     // 22 y cols staged

union V16U { v16bf v; v8bf h[2]; };

__global__ __launch_bounds__(256)
void corr49_wmma_bf16(const float* __restrict__ x,
                      const float* __restrict__ y,
                      float* __restrict__ out) {
    __shared__ bf16_t xs[BH * 16 * KP];          // [hl][m][c]  (8.5 KB)
    __shared__ bf16_t ys[YROWS * YCOLS * KP];    // [ry][col][c] (~47 KB)

    const int w0  = blockIdx.x * 16;
    const int h0  = blockIdx.y * BH;
    const int tid = threadIdx.x;

    // ---- stage x tile: fp32 -> bf16, layout [hl][m][c] ----
    for (int idx = tid; idx < C_DIM * BH * 16; idx += 256) {
        int m  = idx & 15;
        int hl = (idx >> 4) & (BH - 1);
        int c  = idx >> 5;                        // / (BH*16)
        float v = x[(c * H_DIM + h0 + hl) * W_DIM + w0 + m];
        xs[(hl * 16 + m) * KP + c] = (bf16_t)v;
    }
    // ---- stage y tile (zero-padded): layout [ry][col][c] ----
    for (int idx = tid; idx < YROWS * YCOLS * C_DIM; idx += 256) {
        int col = idx % YCOLS;
        int t   = idx / YCOLS;
        int c   = t & (C_DIM - 1);
        int ry  = t >> 7;
        int gh  = h0 - MAXD + ry;
        int gw  = w0 - MAXD + col;
        float v = 0.0f;
        if ((unsigned)gh < (unsigned)H_DIM && (unsigned)gw < (unsigned)W_DIM)
            v = y[(c * H_DIM + gh) * W_DIM + gw];
        ys[(ry * YCOLS + col) * KP + c] = (bf16_t)v;
    }
    __syncthreads();

    const int wave = tid >> 5;       // wave32 on gfx1250
    const int lane = tid & 31;
    const int hf   = lane >> 4;      // half-wave select
    const int n    = lane & 15;      // M index (A) / N index (B)

    // tasks: (dy 0..6) x (hl 0..BH-1)
    for (int task = wave; task < 7 * BH; task += 8) {
        int dy = task >> 1;          // / BH
        int hl = task & (BH - 1);
        int ry = hl + dy;            // staged y row index

        v8f acc0 = {};               // cols w0-3 .. w0+12
        v8f acc1 = {};               // cols w0+3 .. w0+18

        #pragma unroll
        for (int kb = 0; kb < C_DIM; kb += 32) {
            // A (16x32 bf16): lane m=n; K map: k = kb + 16*hf + e (contiguous)
            V16U A;
            const bf16_t* ap = &xs[(hl * 16 + n) * KP + kb + 16 * hf];
            A.h[0] = *(const v8bf*)(ap);
            A.h[1] = *(const v8bf*)(ap + 8);
            // B (32x16 bf16): lane n; K map: k = kb + 16*hf + e (contiguous)
            V16U B0, B1;
            const bf16_t* bp0 = &ys[(ry * YCOLS + n) * KP + kb + 16 * hf];
            B0.h[0] = *(const v8bf*)(bp0);
            B0.h[1] = *(const v8bf*)(bp0 + 8);
            const bf16_t* bp1 = &ys[(ry * YCOLS + n + 6) * KP + kb + 16 * hf];
            B1.h[0] = *(const v8bf*)(bp1);
            B1.h[1] = *(const v8bf*)(bp1 + 8);

            acc0 = __builtin_amdgcn_wmma_f32_16x16x32_bf16(
                       false, A.v, false, B0.v, (short)0, acc0, false, false);
            acc1 = __builtin_amdgcn_wmma_f32_16x16x32_bf16(
                       false, A.v, false, B1.v, (short)0, acc1, false, false);
        }

        // band-predicated scatter of the two 16x16 D tiles
        const int h = h0 + hl;
        #pragma unroll
        for (int r = 0; r < 8; ++r) {
            int mo = r + 8 * hf;                 // output w offset (M row)
            // tile 0: ycol = w0-3+n, dx_idx = n - mo  in [0,6]
            int d0 = n - mo;
            if (d0 >= 0 && d0 <= 6)
                out[((dy * 7 + d0) * H_DIM + h) * W_DIM + w0 + mo] = acc0[r];
            // tile 1: ycol = w0+3+n, dx_idx = n - mo + 6 in [0,6];
            // only the part tile 0 cannot reach (n would exceed 15)
            int d1 = n - mo + 6;
            if (d1 >= 0 && d1 <= 6 && (mo + d1) >= 16)
                out[((dy * 7 + d1) * H_DIM + h) * W_DIM + w0 + mo] = acc1[r];
        }
    }
}

extern "C" void kernel_launch(void* const* d_in, const int* in_sizes, int n_in,
                              void* d_out, int out_size, void* d_ws, size_t ws_size,
                              hipStream_t stream) {
    (void)in_sizes; (void)n_in; (void)d_ws; (void)ws_size; (void)out_size;
    const float* x = (const float*)d_in[0];
    const float* y = (const float*)d_in[1];
    float* out = (float*)d_out;

    dim3 grid(W_DIM / 16, H_DIM / BH);   // 32 x 192
    dim3 block(256);
    corr49_wmma_bf16<<<grid, block, 0, stream>>>(x, y, out);
}
